// RealMambaCore_58162447122866
// MI455X (gfx1250) — matrix operand order repaired and checked by reading, not compiled, not run on hardware
//
#include <hip/hip_runtime.h>

#define D_MODEL 1024
#define STATE_DIM 16
#define BATCH 32
#define SEQ 2048
#define M_ROWS (BATCH * SEQ)                       // 65536
#define PROJ_DIM (2 * D_MODEL + 3 * STATE_DIM)     // 2096
#define KTILE 32
#define NKB (D_MODEL / KTILE)                      // 32 K-steps

typedef __bf16 bf16_t;
typedef bf16_t v16bf __attribute__((ext_vector_type(16)));
typedef bf16_t v8bf  __attribute__((ext_vector_type(8)));
typedef float  v8f   __attribute__((ext_vector_type(8)));

// Double-buffered LDS layout (dynamic shared, base offset 0).
// Each buffer: A tile 128x32 bf16 (8192 B) then B tile (N-major [n][k]).
#define LDS_A_BYTES   (128 * KTILE * 2)            // 8192
#define BUF_STRIDE_BIG (LDS_A_BYTES + 64 * KTILE * 2)   // 8192 + 4096 = 12288
#define BUF_STRIDE_SML (LDS_A_BYTES + 48 * KTILE * 2)   // 8192 + 3072 = 11264

// ---------------------------------------------------------------------------
// CDNA5 async global->LDS copy (ISA 15.18.3, tracked by ASYNCcnt).
// ---------------------------------------------------------------------------
__device__ __forceinline__ void async_load_b128(unsigned lds_off, const void* gptr) {
    asm volatile("global_load_async_to_lds_b128 %0, %1, off"
                 :: "v"(lds_off), "v"(gptr)
                 : "memory");
}
template <int N>
__device__ __forceinline__ void wait_async() {
    asm volatile("s_wait_asynccnt %0" :: "i"(N) : "memory");
}

// ---------------------------------------------------------------------------
// Fragment builders (wave32 WMMA layouts, ISA 7.12.2):
//  A (16x32 bf16, row-striped): lane L -> row m=L&15, half h=L>>4:
//    elements 0..7 = K 8h..8h+7 ; 8..15 = K 16+8h..16+8h+7.
//  B (32x16 bf16, tile kept N-major in LDS): lane L -> col n=L&15:
//    elements 0..7 = K 8h..8h+7 ; 8..15 = K 16+8h..16+8h+7.
// ---------------------------------------------------------------------------
__device__ __forceinline__ v16bf frag_a(const bf16_t* sA, int m, int h) {
    const v8bf a0 = *reinterpret_cast<const v8bf*>(sA + m * KTILE + 8 * h);
    const v8bf a1 = *reinterpret_cast<const v8bf*>(sA + m * KTILE + 16 + 8 * h);
    v16bf a;
#pragma unroll
    for (int i = 0; i < 8; ++i) { a[i] = a0[i]; a[i + 8] = a1[i]; }
    return a;
}
__device__ __forceinline__ v16bf frag_b(const bf16_t* sB, int n, int h) {
    const v8bf b0 = *reinterpret_cast<const v8bf*>(sB + n * KTILE + 8 * h);
    const v8bf b1 = *reinterpret_cast<const v8bf*>(sB + n * KTILE + 16 + 8 * h);
    v16bf b;
#pragma unroll
    for (int i = 0; i < 8; ++i) { b[i] = b0[i]; b[i + 8] = b1[i]; }
    return b;
}

__device__ __forceinline__ float gelu_exact(float v) {
    return 0.5f * v * (1.0f + erff(v * 0.7071067811865476f));
}

// ---------------------------------------------------------------------------
// One-time conversion passes (hoist all f32->bf16 out of the GEMM loops so
// the GEMMs use pure async copies; halves HBM bytes for x).
// ---------------------------------------------------------------------------
__global__ __launch_bounds__(256) void cvt_x_kernel(const float* __restrict__ X,
                                                    bf16_t* __restrict__ Xb) {
    const size_t i = ((size_t)blockIdx.x * 256 + threadIdx.x) * 8;
    const float4* s = reinterpret_cast<const float4*>(X + i);
    const float4 a = s[0], b = s[1];
    v8bf o;
    o[0] = (bf16_t)a.x; o[1] = (bf16_t)a.y; o[2] = (bf16_t)a.z; o[3] = (bf16_t)a.w;
    o[4] = (bf16_t)b.x; o[5] = (bf16_t)b.y; o[6] = (bf16_t)b.z; o[7] = (bf16_t)b.w;
    *reinterpret_cast<v8bf*>(Xb + i) = o;
}

// Transpose-convert: Wt[n][k] = (bf16)W[k*ldw + n], K = D_MODEL. grid.x = N.
__global__ __launch_bounds__(256) void cvt_wT_kernel(const float* __restrict__ W,
                                                     int ldw, bf16_t* __restrict__ Wt) {
    const int nrow = blockIdx.x;
    const int k0 = threadIdx.x * 4;
    bf16_t* d = Wt + (size_t)nrow * D_MODEL + k0;
#pragma unroll
    for (int i = 0; i < 4; ++i)
        d[i] = (bf16_t)W[(size_t)(k0 + i) * ldw + nrow];
}

// ---------------------------------------------------------------------------
// Big GEMM (all-bf16, double-buffered async LDS staging).
// Block tile 128(M) x 64(N), BK=32, 8 waves; wave (wm,wn) owns 32x32 output:
// 4 WMMA per K-step. Each wave issues exactly 3 async copies per K-step, so
// waiting ASYNCcnt<=3 guarantees the previous tile's copies have landed
// while the new ones remain in flight (latency hidden behind WMMA).
// MODE 0: D = gelu(acc+bias) -> bf16 (G). MODE 1: D = Y[row]*acc+bias -> f32.
// ---------------------------------------------------------------------------
template <int MODE>
__global__ __launch_bounds__(256) void gemm_bf16_kernel(
    const bf16_t* __restrict__ A, const bf16_t* __restrict__ Bt,
    const float* __restrict__ bias, const float* __restrict__ Yscale,
    void* __restrict__ Outv) {
    extern __shared__ char smem[];

    const int tid = threadIdx.x;
    const int rowBase = blockIdx.x * 128;
    const int colBase = blockIdx.y * 64;
    const int wid = tid >> 5, lane = tid & 31;
    const int wm = wid & 3, wn = wid >> 2;
    const int n = lane & 15, h = lane >> 4;

    // staging assignments
    const int s_arow = tid >> 1, s_ahalf = tid & 1;  // 128 rows x 2 halves of 32B
    const int s_bn = tid >> 2, s_bq = tid & 3;       // 64 rows x 4 quarters of 16B
    const bf16_t* gaBase = A + (size_t)(rowBase + s_arow) * D_MODEL + s_ahalf * 16;
    const bf16_t* gbBase = Bt + (size_t)(colBase + s_bn) * D_MODEL + s_bq * 8;
    const unsigned laA = (unsigned)(s_arow * 64 + s_ahalf * 32);
    const unsigned laB = (unsigned)(LDS_A_BYTES + s_bn * 64 + s_bq * 16);

    v8f acc00 = {}, acc01 = {}, acc10 = {}, acc11 = {};

    // prologue: stage tile 0 into buffer 0
    async_load_b128(laA, gaBase);
    async_load_b128(laA + 16, gaBase + 8);
    async_load_b128(laB, gbBase);

    for (int kb = 0; kb < NKB; ++kb) {
        const unsigned curBuf = (kb & 1) * BUF_STRIDE_BIG;
        if (kb + 1 < NKB) {
            const unsigned nxtBuf = ((kb + 1) & 1) * BUF_STRIDE_BIG;
            const bf16_t* ga = gaBase + (kb + 1) * KTILE;
            async_load_b128(nxtBuf + laA, ga);
            async_load_b128(nxtBuf + laA + 16, ga + 8);
            async_load_b128(nxtBuf + laB, gbBase + (kb + 1) * KTILE);
            if (kb + 2 < NKB)
                __builtin_prefetch((const void*)(ga + KTILE), 0, 1);  // global_prefetch_b8
            wait_async<3>();   // previous tile complete; new 3 still in flight
        } else {
            wait_async<0>();
        }
        __syncthreads();

        const bf16_t* sA = reinterpret_cast<const bf16_t*>(smem + curBuf);
        const bf16_t* sB = reinterpret_cast<const bf16_t*>(smem + curBuf + LDS_A_BYTES);
        {
            v16bf a0 = frag_a(sA, wm * 32 + n, h);
            v16bf a1 = frag_a(sA, wm * 32 + 16 + n, h);
            v16bf b0 = frag_b(sB, wn * 32 + n, h);
            v16bf b1 = frag_b(sB, wn * 32 + 16 + n, h);
            acc00 = __builtin_amdgcn_wmma_f32_16x16x32_bf16(false, a0, false, b0, (short)0, acc00, false, false);
            acc01 = __builtin_amdgcn_wmma_f32_16x16x32_bf16(false, a0, false, b1, (short)0, acc01, false, false);
            acc10 = __builtin_amdgcn_wmma_f32_16x16x32_bf16(false, a1, false, b0, (short)0, acc10, false, false);
            acc11 = __builtin_amdgcn_wmma_f32_16x16x32_bf16(false, a1, false, b1, (short)0, acc11, false, false);
        }
        __syncthreads();   // buf reused two iterations later; reads are done (consumed by WMMA)
    }

    const int c0 = colBase + wn * 32 + n;
    const int c1 = c0 + 16;
#pragma unroll
    for (int j = 0; j < 8; ++j) {
        const int m0 = wm * 32 + j + 8 * h;
        const int m1 = m0 + 16;
        const size_t r0 = (size_t)rowBase + m0;
        const size_t r1 = (size_t)rowBase + m1;
        if (MODE == 0) {
            bf16_t* G = (bf16_t*)Outv;
            G[r0 * D_MODEL + c0] = (bf16_t)gelu_exact(acc00[j] + bias[c0]);
            G[r0 * D_MODEL + c1] = (bf16_t)gelu_exact(acc01[j] + bias[c1]);
            G[r1 * D_MODEL + c0] = (bf16_t)gelu_exact(acc10[j] + bias[c0]);
            G[r1 * D_MODEL + c1] = (bf16_t)gelu_exact(acc11[j] + bias[c1]);
        } else {
            float* Out = (float*)Outv;
            const float y0 = Yscale[r0], y1 = Yscale[r1];
            Out[r0 * D_MODEL + c0] = y0 * acc00[j] + bias[c0];
            Out[r0 * D_MODEL + c1] = y0 * acc01[j] + bias[c1];
            Out[r1 * D_MODEL + c0] = y1 * acc10[j] + bias[c0];
            Out[r1 * D_MODEL + c1] = y1 * acc11[j] + bias[c1];
        }
    }
}

// ---------------------------------------------------------------------------
// Small GEMM: P[M,48] = Xb @ WsT^T + bias (B|C|delta_raw columns).
// Block tile 128(M) x 48(N); 8 waves, wave = 16-row tile, 3 WMMA per K-step.
// B staging mapped tid%192 (threads 192..255 duplicate tasks 0..63) so every
// wave issues exactly 3 async copies per K-step -> ASYNCcnt<=3 wait is exact.
// ---------------------------------------------------------------------------
__global__ __launch_bounds__(256) void gemm_proj_s_kernel(
    const bf16_t* __restrict__ Xb, const bf16_t* __restrict__ WsT,
    const float* __restrict__ bias, float* __restrict__ P) {
    extern __shared__ char smem[];

    const int tid = threadIdx.x;
    const int rowBase = blockIdx.x * 128;
    const int wid = tid >> 5, lane = tid & 31;
    const int n = lane & 15, h = lane >> 4;

    const int s_arow = tid >> 1, s_ahalf = tid & 1;
    const int btask = (tid < 192) ? tid : (tid - 192);
    const int s_bn = btask >> 2, s_bq = btask & 3;
    const bf16_t* gaBase = Xb + (size_t)(rowBase + s_arow) * D_MODEL + s_ahalf * 16;
    const bf16_t* gbBase = WsT + (size_t)s_bn * D_MODEL + s_bq * 8;
    const unsigned laA = (unsigned)(s_arow * 64 + s_ahalf * 32);
    const unsigned laB = (unsigned)(LDS_A_BYTES + s_bn * 64 + s_bq * 16);

    v8f acc[3] = {};

    async_load_b128(laA, gaBase);
    async_load_b128(laA + 16, gaBase + 8);
    async_load_b128(laB, gbBase);

    for (int kb = 0; kb < NKB; ++kb) {
        const unsigned curBuf = (kb & 1) * BUF_STRIDE_SML;
        if (kb + 1 < NKB) {
            const unsigned nxtBuf = ((kb + 1) & 1) * BUF_STRIDE_SML;
            const bf16_t* ga = gaBase + (kb + 1) * KTILE;
            async_load_b128(nxtBuf + laA, ga);
            async_load_b128(nxtBuf + laA + 16, ga + 8);
            async_load_b128(nxtBuf + laB, gbBase + (kb + 1) * KTILE);
            wait_async<3>();
        } else {
            wait_async<0>();
        }
        __syncthreads();

        const bf16_t* sA = reinterpret_cast<const bf16_t*>(smem + curBuf);
        const bf16_t* sB = reinterpret_cast<const bf16_t*>(smem + curBuf + LDS_A_BYTES);
        {
            v16bf a = frag_a(sA, wid * 16 + n, h);
#pragma unroll
            for (int t = 0; t < 3; ++t) {
                v16bf b = frag_b(sB, t * 16 + n, h);
                acc[t] = __builtin_amdgcn_wmma_f32_16x16x32_bf16(false, a, false, b,
                                                                 (short)0, acc[t], false, false);
            }
        }
        __syncthreads();
    }

#pragma unroll
    for (int t = 0; t < 3; ++t) {
#pragma unroll
        for (int j = 0; j < 8; ++j) {
            const int m = wid * 16 + j + 8 * h;
            const size_t row = (size_t)rowBase + m;
            const int c = t * 16 + n;
            P[row * 48 + c] = acc[t][j] + bias[c];
        }
    }
}

// ---------------------------------------------------------------------------
// Selective scan: one wave per batch, lane = state dim (0..15).
// ---------------------------------------------------------------------------
__global__ __launch_bounds__(32) void scan_kernel(
    const float* __restrict__ P, const float* __restrict__ A_log,
    float* __restrict__ Y) {
    const int b = blockIdx.x;
    const int lane = threadIdx.x;
    const int n = lane & 15;
    const float A = -__expf(A_log[n]);
    float state = 0.0f;
    for (int s = 0; s < SEQ; ++s) {
        const float* p = P + ((size_t)b * SEQ + s) * 48;
        const float Bt = p[n];
        const float Ct = p[16 + n];
        const float dr = p[32 + n];
        const float delta = (dr > 20.0f) ? dr : log1pf(__expf(dr));
        state = __expf(A * delta) * state + Bt;
        float prod = (lane < 16) ? state * Ct : 0.0f;
        prod += __shfl_down(prod, 8, 32);
        prod += __shfl_down(prod, 4, 32);
        prod += __shfl_down(prod, 2, 32);
        prod += __shfl_down(prod, 1, 32);
        if (lane == 0) Y[(size_t)b * SEQ + s] = prod;
    }
}

// ---------------------------------------------------------------------------
extern "C" void kernel_launch(void* const* d_in, const int* in_sizes, int n_in,
                              void* d_out, int out_size, void* d_ws, size_t ws_size,
                              hipStream_t stream) {
    const float* x     = (const float*)d_in[0];
    const float* W_in  = (const float*)d_in[1];
    const float* b_in  = (const float*)d_in[2];
    const float* A_log = (const float*)d_in[3];
    const float* W_out = (const float*)d_in[4];
    const float* b_out = (const float*)d_in[5];
    float* out = (float*)d_out;

    char* ws = (char*)d_ws;
    const size_t XB_BYTES  = (size_t)M_ROWS * D_MODEL * sizeof(bf16_t);   // 128 MiB
    const size_t G_BYTES   = (size_t)M_ROWS * D_MODEL * sizeof(bf16_t);   // 128 MiB
    const size_t P_BYTES   = (size_t)M_ROWS * 48 * sizeof(float);         // 12 MiB
    const size_t Y_BYTES   = (size_t)M_ROWS * sizeof(float);
    const size_t WZT_BYTES = (size_t)D_MODEL * D_MODEL * sizeof(bf16_t);  // 2 MiB
    const size_t WST_BYTES = (size_t)48 * D_MODEL * sizeof(bf16_t);

    bf16_t* Xb  = (bf16_t*)ws;                                       ws += XB_BYTES;
    bf16_t* G   = (bf16_t*)ws;                                       ws += G_BYTES;
    float*  P   = (float*)ws;                                        ws += P_BYTES;
    float*  Y   = (float*)ws;                                        ws += Y_BYTES;
    bf16_t* WzT = (bf16_t*)ws;                                       ws += WZT_BYTES;
    bf16_t* WsT = (bf16_t*)ws;                                       ws += WST_BYTES;
    bf16_t* WoT = (bf16_t*)ws;

    // one-time bf16 conversion / weight transposes
    cvt_x_kernel<<<(M_ROWS * D_MODEL) / (256 * 8), 256, 0, stream>>>(x, Xb);
    cvt_wT_kernel<<<D_MODEL, 256, 0, stream>>>(W_in + D_MODEL, PROJ_DIM, WzT);
    cvt_wT_kernel<<<48, 256, 0, stream>>>(W_in + 2 * D_MODEL, PROJ_DIM, WsT);
    cvt_wT_kernel<<<D_MODEL, 256, 0, stream>>>(W_out, D_MODEL, WoT);

    const dim3 gridBig(M_ROWS / 128, D_MODEL / 64);
    const size_t ldsBig = 2 * BUF_STRIDE_BIG;   // 24576 B
    const size_t ldsSml = 2 * BUF_STRIDE_SML;   // 22528 B

    // z-projection + gelu -> bf16 G
    gemm_bf16_kernel<0><<<gridBig, 256, ldsBig, stream>>>(Xb, WzT, b_in + D_MODEL, nullptr, (void*)G);
    // B/C/delta projection -> f32 P
    gemm_proj_s_kernel<<<M_ROWS / 128, 256, ldsSml, stream>>>(Xb, WsT, b_in + 2 * D_MODEL, P);
    // recurrence -> Y
    scan_kernel<<<BATCH, 32, 0, stream>>>(P, A_log, Y);
    // out = Y*(G @ W_out) + b_out
    gemm_bf16_kernel<1><<<gridBig, 256, ldsBig, stream>>>(G, WoT, b_out, Y, (void*)out);
}
